// SelfAttention_39625368273340
// MI455X (gfx1250) — matrix operand order repaired and checked by reading, not compiled, not run on hardware
//
#include <hip/hip_runtime.h>
#include <hip/hip_bf16.h>

#define BB 4
#define CC 256
#define DD 32
#define NN 4096
#define KT 64                       // keys per pipeline stage in k_attn
#define QK_SCALE 0.17677669529663687f  // 1/sqrt(D)

typedef unsigned short u16;
typedef u16    v8u16  __attribute__((ext_vector_type(8)));
typedef u16    v16u16 __attribute__((ext_vector_type(16)));
typedef __bf16 bf16_t;
typedef bf16_t v16bf  __attribute__((ext_vector_type(16)));
typedef float  v8f    __attribute__((ext_vector_type(8)));

static __device__ __forceinline__ u16 f2bf(float f) {
  unsigned u = __float_as_uint(f);
  u += 0x7FFFu + ((u >> 16) & 1u);   // round-to-nearest-even
  return (u16)(u >> 16);
}

static __device__ __forceinline__ v16bf as_bf16x16(v16u16 x) {
  v16bf r;
  __builtin_memcpy(&r, &x, sizeof(r));
  return r;
}

static __device__ __forceinline__ v8f vzero8() {
  v8f z;
  #pragma unroll
  for (int i = 0; i < 8; ++i) z[i] = 0.0f;
  return z;
}

// A-fragment 16x32 bf16 (M = lane%16). Per ISA layout: elements 0..7 hold
// K = kbA..kbA+7, elements 8..15 hold K = kbA+16..kbA+23, kbA = (lane<16?0:8).
static __device__ __forceinline__ v16bf load_a(const u16* p, int kbA) {
  v8u16 lo = *(const v8u16*)(p + kbA);
  v8u16 hi = *(const v8u16*)(p + kbA + 16);
  v16u16 c = __builtin_shufflevector(lo, hi, 0,1,2,3,4,5,6,7,8,9,10,11,12,13,14,15);
  return as_bf16x16(c);
}

// B-fragment 32x16 bf16, storage [n][k] (k contiguous). Lane holds column
// n = lane%16, elements 0..7 = K kbB/2..kbB/2+7, 8..15 = K kbB/2+16..kbB/2+23, kbB = (lane<16?0:16).
static __device__ __forceinline__ v16bf load_b(const u16* p, int kbB) {
  v8u16 lo = *(const v8u16*)(p + kbB / 2);
  v8u16 hi = *(const v8u16*)(p + kbB / 2 + 16);
  v16u16 c = __builtin_shufflevector(lo, hi, 0,1,2,3,4,5,6,7,8,9,10,11,12,13,14,15);
  return as_bf16x16(c);
}

static __device__ __forceinline__ v8f wmma_bf(v16bf a, v16bf b, v8f c) {
  return __builtin_amdgcn_wmma_f32_16x16x32_bf16(false, a, false, b, (short)0, c,
                                                 false, false);
}

// 16-byte async copy global -> LDS (CDNA5, tracked by ASYNCcnt)
static __device__ __forceinline__ void async_cp16(unsigned lds_byte,
                                                  unsigned long long gaddr) {
  asm volatile("global_load_async_to_lds_b128 %0, %1, off"
               :: "v"(lds_byte), "v"(gaddr) : "memory");
}

// ---------------- phase 0: layout conversion -------------------------------

__global__ void k_convert_x(const float* __restrict__ x, u16* __restrict__ xt) {
  int t = blockIdx.x * 256 + threadIdx.x;          // t = ((b*N + n)*C + c)
  int c = t & (CC - 1);
  int n = (t >> 8) & (NN - 1);
  int b = t >> 20;
  xt[t] = f2bf(x[((size_t)(b * CC + c)) * NN + n]);
}

__global__ void k_convert_w(const float* __restrict__ Wq, const float* __restrict__ Wk,
                            const float* __restrict__ Wv, const float* __restrict__ Wp,
                            u16* __restrict__ wqb, u16* __restrict__ wkb,
                            u16* __restrict__ wvb, u16* __restrict__ wpb) {
  int t = blockIdx.x * 256 + threadIdx.x;
  if (t < DD * CC) wqb[t] = f2bf(Wq[t] * QK_SCALE);          // fold softmax scale
  int t1 = t - DD * CC;
  if (t1 >= 0 && t1 < DD * CC) wkb[t1] = f2bf(Wk[t1]);
  int t2 = t - 2 * DD * CC;
  if (t2 >= 0 && t2 < CC * CC) wvb[t2] = f2bf(Wv[t2]);
  int t3 = t - (2 * DD * CC + CC * CC);
  if (t3 >= 0 && t3 < CC * CC) wpb[t3] = f2bf(Wp[t3]);
}

// ---------------- phase 1: q/k/v projections -------------------------------

__global__ void __launch_bounds__(256)
k_proj_qkv(const u16* __restrict__ xt,
           const u16* __restrict__ wqb, const u16* __restrict__ wkb,
           const u16* __restrict__ wvb,
           const float* __restrict__ bq, const float* __restrict__ bk,
           const float* __restrict__ bv,
           u16* __restrict__ qb, u16* __restrict__ kb, u16* __restrict__ vb) {
  const int lane = threadIdx.x & 31;
  const int wave = threadIdx.x >> 5;
  const int b    = blockIdx.x >> 5;     // N/128 = 32 pixel-tiles per batch
  const int tile = blockIdx.x & 31;
  const int col  = lane & 15;
  const int half = lane >> 4;
  const int kbA  = half * 8;
  const int kbB  = half * 16;
  const int pix0 = tile * 128 + wave * 16;
  const int m    = pix0 + col;

  const u16* xrow = xt + ((size_t)(b * NN + m)) * CC;

  // ---- Q and K (2 d-tiles each) ----
  v8f qa[2], ka[2];
  qa[0] = qa[1] = ka[0] = ka[1] = vzero8();
  #pragma unroll
  for (int ks = 0; ks < 8; ++ks) {
    v16bf a = load_a(xrow + ks * 32, kbA);
    #pragma unroll
    for (int dt = 0; dt < 2; ++dt) {
      qa[dt] = wmma_bf(a, load_b(wqb + (size_t)(dt * 16 + col) * CC + ks * 32, kbB), qa[dt]);
      ka[dt] = wmma_bf(a, load_b(wkb + (size_t)(dt * 16 + col) * CC + ks * 32, kbB), ka[dt]);
    }
  }
  #pragma unroll
  for (int dt = 0; dt < 2; ++dt) {
    int d = dt * 16 + col;
    float bqv = bq[d] * QK_SCALE;
    float bkv = bk[d];
    #pragma unroll
    for (int r = 0; r < 8; ++r) {
      int pix = pix0 + r + half * 8;
      qb[((size_t)(b * NN + pix)) * DD + d] = f2bf(qa[dt][r] + bqv);
      kb[((size_t)(b * NN + pix)) * DD + d] = f2bf(ka[dt][r] + bkv);
    }
  }

  // ---- V (16 c-tiles) ----
  v8f va[16];
  #pragma unroll
  for (int i = 0; i < 16; ++i) va[i] = vzero8();
  #pragma unroll
  for (int ks = 0; ks < 8; ++ks) {
    v16bf a = load_a(xrow + ks * 32, kbA);
    #pragma unroll
    for (int ct = 0; ct < 16; ++ct)
      va[ct] = wmma_bf(a, load_b(wvb + (size_t)(ct * 16 + col) * CC + ks * 32, kbB), va[ct]);
  }
  #pragma unroll
  for (int ct = 0; ct < 16; ++ct) {
    int c = ct * 16 + col;
    float bvv = bv[c];
    #pragma unroll
    for (int r = 0; r < 8; ++r) {
      int pix = pix0 + r + half * 8;
      vb[((size_t)(b * CC + c)) * NN + pix] = f2bf(va[ct][r] + bvv);
    }
  }
}

// ---------------- phase 2: flash attention + fused output projection -------
// 4 waves/WG, each owning a 16-query tile; the WG shares a double-buffered
// K/V stage in LDS filled with global_load_async_to_lds_b128 (ASYNCcnt) so
// each K/V tile is fetched once per WG and the copy overlaps WMMA compute.

// LDS element (u16) offsets
#define SM_K(buf)  ((buf) * (KT * DD))                 // 2 x 2048
#define SM_V(buf)  (2 * KT * DD + (buf) * (CC * KT))   // 2 x 16384
#define SM_P(w)    (2 * KT * DD + 2 * CC * KT + (w) * (16 * KT))
#define SM_O(w)    (SM_V(0) + (w) * (16 * CC))         // aliases dead V buffers
#define SM_TOTAL   (2 * KT * DD + 2 * CC * KT + 4 * 16 * KT)   // 40960 u16 = 80 KB

__global__ void __launch_bounds__(128)
k_attn(const u16* __restrict__ qb, const u16* __restrict__ kb,
       const u16* __restrict__ vb, const u16* __restrict__ wpb,
       const float* __restrict__ bp, const float* __restrict__ gamma,
       const float* __restrict__ x, float* __restrict__ out) {
  __shared__ __align__(32) u16 smem[SM_TOTAL];

  const int lane = threadIdx.x & 31;
  const int wave = threadIdx.x >> 5;
  const int col  = lane & 15;
  const int half = lane >> 4;
  const int kbA  = half * 8;
  const int kbB  = half * 16;
  const int roff = half * 8;
  const int b    = blockIdx.x >> 6;   // N/64 query-groups per batch
  const int qg   = blockIdx.x & 63;
  const int qt   = qg * 4 + wave;     // this wave's 16-query tile

  const unsigned lds_base = (unsigned)(size_t)(void*)smem;  // flat low 32b = LDS addr
  const char* kglob = (const char*)(kb + (size_t)b * NN * DD);
  const char* vglob = (const char*)(vb + (size_t)b * CC * NN);

  // Each wave issues its share of a 64-key stage: K = 4 KB contiguous,
  // V = 256 rows x 128 B (global row stride N*2).
  auto issue_tile = [&](int key0, int buf) {
    const char* kg = kglob + (size_t)key0 * DD * 2;
    unsigned    kl = lds_base + SM_K(buf) * 2;
    #pragma unroll
    for (int i = 0; i < 2; ++i) {
      int n = wave * 64 + i * 32 + lane;                     // 16B chunk id, 0..255
      async_cp16(kl + n * 16, (unsigned long long)(size_t)(kg + n * 16));
    }
    const char* vg = vglob + (size_t)key0 * 2;
    unsigned    vl = lds_base + SM_V(buf) * 2;
    #pragma unroll
    for (int i = 0; i < 16; ++i) {
      int n = wave * 512 + i * 32 + lane;                    // chunk id, 0..2047
      int c = n >> 3, j = n & 7;
      async_cp16(vl + n * 16,
                 (unsigned long long)(size_t)(vg + (size_t)c * (NN * 2) + j * 16));
    }
  };

  // Q A-fragment: K = D = 32 fits one WMMA exactly.
  v16bf aq = load_a(qb + ((size_t)(b * NN + qt * 16 + col)) * DD, kbA);

  v8f acc[16];
  #pragma unroll
  for (int i = 0; i < 16; ++i) acc[i] = vzero8();
  float Mr[8], Lr[8];
  #pragma unroll
  for (int r = 0; r < 8; ++r) { Mr[r] = -1e30f; Lr[r] = 0.0f; }

  u16* Pb = smem + SM_P(wave);

  issue_tile(0, 0);
  asm volatile("s_wait_asynccnt 0x0" ::: "memory");
  __syncthreads();

  for (int kt = 0; kt < NN / KT; ++kt) {
    const int buf = kt & 1;
    if (kt + 1 < NN / KT) issue_tile((kt + 1) * KT, buf ^ 1);

    const u16* Kb = smem + SM_K(buf);
    const u16* Vb = smem + SM_V(buf);

    // S = q @ k^T over 64 keys: 4 WMMA tiles out of LDS
    v8f s[4];
    #pragma unroll
    for (int j = 0; j < 4; ++j)
      s[j] = wmma_bf(aq, load_b(Kb + (j * 16 + col) * DD, kbB), vzero8());

    // online softmax
    float f[8];
    bool anyscale = false;
    #pragma unroll
    for (int r = 0; r < 8; ++r) {
      float mr = fmaxf(fmaxf(s[0][r], s[1][r]), fmaxf(s[2][r], s[3][r]));
      mr = fmaxf(mr, __shfl_xor(mr, 1, 16));
      mr = fmaxf(mr, __shfl_xor(mr, 2, 16));
      mr = fmaxf(mr, __shfl_xor(mr, 4, 16));
      mr = fmaxf(mr, __shfl_xor(mr, 8, 16));
      float nm = fmaxf(Mr[r], mr);
      f[r] = __expf(Mr[r] - nm);
      anyscale |= (f[r] < 1.0f);
      Mr[r] = nm;
      int row = r + roff;
      float rs = 0.0f;
      #pragma unroll
      for (int j = 0; j < 4; ++j) {
        float p = __expf(s[j][r] - nm);
        rs += p;
        Pb[row * KT + j * 16 + col] = f2bf(p);
      }
      rs += __shfl_xor(rs, 1, 16);
      rs += __shfl_xor(rs, 2, 16);
      rs += __shfl_xor(rs, 4, 16);
      rs += __shfl_xor(rs, 8, 16);
      Lr[r] = Lr[r] * f[r] + rs;
    }
    // rescale accumulator only when the running max actually moved
    if (__ballot(anyscale)) {
      #pragma unroll
      for (int ct = 0; ct < 16; ++ct)
        #pragma unroll
        for (int r = 0; r < 8; ++r) acc[ct][r] *= f[r];
    }

    // same-wave LDS fence: P stores above land before cross-lane reads
    asm volatile("s_wait_dscnt 0x0" ::: "memory");
    v16bf ap0 = load_a(Pb + col * KT, kbA);
    v16bf ap1 = load_a(Pb + col * KT + 32, kbA);

    #pragma unroll
    for (int ct = 0; ct < 16; ++ct) {
      const u16* vrow = Vb + (ct * 16 + col) * KT;
      acc[ct] = wmma_bf(ap0, load_b(vrow, kbB), acc[ct]);
      acc[ct] = wmma_bf(ap1, load_b(vrow + 32, kbB), acc[ct]);
    }

    // next stage must be resident (own ASYNCcnt) and WG-visible (barrier)
    asm volatile("s_wait_asynccnt 0x0" ::: "memory");
    __syncthreads();
  }

  // normalize and stash O (pixel-major, channel contiguous) in LDS
  u16* Ob = smem + SM_O(wave);           // V buffers are dead past the barrier
  #pragma unroll
  for (int r = 0; r < 8; ++r) {
    float inv = 1.0f / Lr[r];
    #pragma unroll
    for (int ct = 0; ct < 16; ++ct)
      Ob[(r + roff) * CC + ct * 16 + col] = f2bf(acc[ct][r] * inv);
  }
  asm volatile("s_wait_dscnt 0x0" ::: "memory");

  // fused output projection: out = x + gamma * (Wp @ O + bp)
  v8f oacc[16];
  #pragma unroll
  for (int i = 0; i < 16; ++i) oacc[i] = vzero8();
  #pragma unroll
  for (int ks = 0; ks < 8; ++ks) {
    v16bf ao = load_a(Ob + col * CC + ks * 32, kbA);
    #pragma unroll
    for (int ot = 0; ot < 16; ++ot) {
      v16bf bw = load_b(wpb + (size_t)(ot * 16 + col) * CC + ks * 32, kbB);
      oacc[ot] = wmma_bf(ao, bw, oacc[ot]);
    }
  }
  const float g = gamma[0];
  #pragma unroll
  for (int ot = 0; ot < 16; ++ot) {
    int cout = ot * 16 + col;
    float bpv = bp[cout];
    #pragma unroll
    for (int r = 0; r < 8; ++r) {
      int pix = qt * 16 + r + roff;
      size_t oi = ((size_t)(b * CC + cout)) * NN + pix;
      out[oi] = x[oi] + g * (oacc[ot][r] + bpv);
    }
  }
}

// ---------------- launch ---------------------------------------------------

extern "C" void kernel_launch(void* const* d_in, const int* in_sizes, int n_in,
                              void* d_out, int out_size, void* d_ws, size_t ws_size,
                              hipStream_t stream) {
  const float* x  = (const float*)d_in[0];
  const float* Wq = (const float*)d_in[1];
  const float* bq = (const float*)d_in[2];
  const float* Wk = (const float*)d_in[3];
  const float* bk = (const float*)d_in[4];
  const float* Wv = (const float*)d_in[5];
  const float* bv = (const float*)d_in[6];
  const float* Wp = (const float*)d_in[7];
  const float* bp = (const float*)d_in[8];
  const float* gm = (const float*)d_in[9];
  float* out = (float*)d_out;

  char* ws = (char*)d_ws;
  size_t off = 0;
  u16* xt  = (u16*)(ws + off); off += (size_t)BB * NN * CC * 2;  // 8.39 MB
  u16* qb  = (u16*)(ws + off); off += (size_t)BB * NN * DD * 2;  // 1.05 MB
  u16* kb  = (u16*)(ws + off); off += (size_t)BB * NN * DD * 2;  // 1.05 MB
  u16* vb  = (u16*)(ws + off); off += (size_t)BB * CC * NN * 2;  // 8.39 MB
  u16* wqb = (u16*)(ws + off); off += (size_t)DD * CC * 2;
  u16* wkb = (u16*)(ws + off); off += (size_t)DD * CC * 2;
  u16* wvb = (u16*)(ws + off); off += (size_t)CC * CC * 2;
  u16* wpb = (u16*)(ws + off); off += (size_t)CC * CC * 2;

  k_convert_x<<<(BB * CC * NN) / 256, 256, 0, stream>>>(x, xt);
  k_convert_w<<<(2 * DD * CC + 2 * CC * CC + 255) / 256, 256, 0, stream>>>(
      Wq, Wk, Wv, Wp, wqb, wkb, wvb, wpb);
  k_proj_qkv<<<BB * (NN / 128), 256, 0, stream>>>(xt, wqb, wkb, wvb, bq, bk, bv,
                                                  qb, kb, vb);
  k_attn<<<BB * (NN / KT / 4) * 4, 128, 0, stream>>>(qb, kb, vb, wpb, bp, gm, x, out);
}
